// PerceiverResampler_25237227832051
// MI455X (gfx1250) — matrix operand order repaired and checked by reading, not compiled, not run on hardware
//
#include <hip/hip_runtime.h>
#include <hip/hip_bf16.h>

// ---------------------------------------------------------------------------
// PerceiverResampler on MI455X (gfx1250): bf16 WMMA GEMMs + flash attention.
// GEMM staging uses CDNA5 async copy paths:
//   - A tiles: global_load_async_to_lds_b128 (ASYNCcnt)
//   - B tiles: Tensor Data Mover tensor_load_to_lds (TENSORcnt), 2-D descriptor
// with a 2-deep LDS double buffer so staging overlaps WMMA issue.
// ---------------------------------------------------------------------------

typedef __attribute__((ext_vector_type(16))) __bf16 bf16x16;
typedef __attribute__((ext_vector_type(8)))  __bf16 bf16x8;
typedef __attribute__((ext_vector_type(8)))  float  f32x8;
typedef __attribute__((ext_vector_type(4)))  unsigned int u32x4;
typedef __attribute__((ext_vector_type(8)))  int i32x8;
typedef __attribute__((ext_vector_type(4)))  int i32x4;

union BF16x16 { bf16x16 v; bf16x8 h[2]; };

static __device__ __forceinline__ f32x8 fzero8() {
  f32x8 v; for (int i = 0; i < 8; ++i) v[i] = 0.f; return v;
}
static __device__ __forceinline__ f32x8 wmma_bf16(bf16x16 a, bf16x16 b, f32x8 c) {
  // D = A(16x32) * B(32x16) + C, fp32 accumulate
  return __builtin_amdgcn_wmma_f32_16x16x32_bf16(false, a, false, b, (short)0, c, false, false);
}

// Per-lane async 16B copy global -> LDS (ASYNCcnt). Flat LDS addresses carry
// the LDS byte offset in addr[31:0], so truncating the generic pointer is the
// documented LDS address for the VDST operand.
static __device__ __forceinline__ void async_b128(const void* g, void* l) {
  const unsigned lds = (unsigned)(unsigned long long)l;
  const unsigned long long ga = (unsigned long long)g;
  asm volatile("global_load_async_to_lds_b128 %0, %1, off"
               :: "v"(lds), "v"(ga) : "memory");
}
static __device__ __forceinline__ void wait_async0() {
  asm volatile("s_wait_asynccnt 0" ::: "memory");
}

// TDM: load one 32x64 bf16 tile (rows strided by ldb elements) into LDS with
// 16B padding inserted every 128B so rows land at a 144B stride (bank-spread).
// D# group layout per CDNA5 ISA. This toolchain exposes
// the 6-arg builtin: (u32x4 g0, i32x8 g1, i32x4 g2, i32x4 g3, i32x8 g4, cpol).
static __device__ __forceinline__ void tdm_load_b_tile(
    const __bf16* gsrc, int ldb, void* ldst) {
  const unsigned long long ga = (unsigned long long)gsrc;
  const unsigned lds = (unsigned)(unsigned long long)ldst;
  u32x4 g0;
  g0[0] = 1u;                                       // count=1, user descriptor
  g0[1] = lds;                                      // lds_addr (bytes)
  g0[2] = (unsigned)(ga & 0xffffffffull);           // global_addr[31:0]
  g0[3] = (unsigned)((ga >> 32) & 0x01ffffffull)    // global_addr[56:32]
        | (2u << 30);                               // type = 2 ("image")
  i32x8 g1;
  g1[0] = (int)((1u << 16)                          // data_size = 2 bytes
        |       (1u << 20)                          // pad_enable
        |       (4u << 22)                          // pad_interval: every 32 DW (128B)
        |       (3u << 25));                        // pad_amount: 4 DW (16B)
  g1[1] = (int)(64u << 16);                         // tensor_dim0 = 64 (lo16)
  g1[2] = (int)(32u << 16);                         // dim0 hi=0 | tensor_dim1 = 32 (lo16)
  g1[3] = (int)(64u << 16);                         // dim1 hi=0 | tile_dim0 = 64
  g1[4] = (int)32u;                                 // tile_dim1 = 32, tile_dim2 = 0
  g1[5] = ldb;                                      // tensor_dim0_stride[31:0] (elements)
  g1[6] = 0;                                        // stride hi | tensor_dim1_stride lo
  g1[7] = 0;
  i32x4 g2 = {0, 0, 0, 0};                          // unused (2-D tile)
  i32x4 g3 = {0, 0, 0, 0};
  i32x8 g4 = {0, 0, 0, 0, 0, 0, 0, 0};              // unused trailing group
  __builtin_amdgcn_tensor_load_to_lds(g0, g1, g2, g3, g4, 0);
}

// Problem constants
#define cB  16
#define cN  2048
#define cD  1024
#define cH  8
#define cDH 64
#define cI  512
#define cL  64
#define cDEPTH 6
#define cFF 4096
#define cS  2112   // N + L

// Epilogue modes for the GEMM kernel
#define EPI_F32_RESID 0   // outF = acc + resF (fp32)
#define EPI_GELU_BF16 1   // outH = bf16(gelu(acc))
#define EPI_KV        2   // scatter into kT (B,H,DH,S) and v (B,H,S,DH), bf16
#define EPI_QH        3   // scatter into q  (B,H,L,DH), bf16

// ---------------------------------------------------------------------------
// Tiled bf16 GEMM: C[M,N] = A[M,K] * B[K,N], 64x64 block tile, 8 waves,
// double-buffered async staging. M and N are multiples of 64 for every call.
// ---------------------------------------------------------------------------
__global__ __launch_bounds__(256) void gemm_bf16_wmma(
    const __bf16* __restrict__ A, int lda,
    const __bf16* __restrict__ Bw, int ldb,
    int M, int Nc, int K, int epi, int ldc,
    float* __restrict__ outF, const float* __restrict__ resF,
    __bf16* __restrict__ outH,
    __bf16* __restrict__ kOut, __bf16* __restrict__ vOut, int Sseq)
{
  __shared__ __bf16 As[2][64][40];  // 80 B row stride (16B-aligned)
  __shared__ __bf16 Bs[2][32][72];  // 144 B row stride (16B-aligned)

  const int tid  = threadIdx.x;
  const int wid  = tid >> 5;
  const int lane = tid & 31;
  const int wm   = wid & 3;
  const int wn   = wid >> 2;
  const int m0   = blockIdx.y * 64;
  const int n0   = blockIdx.x * 64;
  const int half = lane >> 4;
  const int r16  = lane & 15;

  // Staging coordinates (one 16B async copy per thread for the A tile)
  const int rA  = tid >> 2;
  const int cgA = (tid & 3) * 8;

  f32x8 acc0 = fzero8();
  f32x8 acc1 = fzero8();

  const int nk = K >> 5;   // K is a multiple of 32

  // Prologue: stage tiles for k-step 0 into buffer 0
  {
    const __bf16* ap = A + (size_t)(m0 + rA) * lda + cgA;
    async_b128(ap, &As[0][rA][cgA]);
    __builtin_prefetch(ap + 32, 0, 1);              // global_prefetch_b8 of next K tile
    if (wid == 0)
      tdm_load_b_tile(Bw + n0, ldb, &Bs[0][0][0]);
  }
  wait_async0();
  __builtin_amdgcn_s_wait_tensorcnt(0);
  __syncthreads();

  for (int ki = 0; ki < nk; ++ki) {
    const int cur = ki & 1, nxt = cur ^ 1;
    // Stage next K tile while computing the current one
    if (ki + 1 < nk) {
      const int k1 = (ki + 1) << 5;
      const __bf16* ap = A + (size_t)(m0 + rA) * lda + k1 + cgA;
      async_b128(ap, &As[nxt][rA][cgA]);
      __builtin_prefetch(ap + 32, 0, 1);
      if (wid == 0)
        tdm_load_b_tile(Bw + (size_t)k1 * ldb + n0, ldb, &Bs[nxt][0][0]);
    }

    // A fragment: lane row = 16*wm + r16, K runs [8h,8h+8) and [16+8h,16+8h+8)
    BF16x16 af;
    {
      const __bf16* ar = &As[cur][16 * wm + r16][0];
      af.h[0] = *(const bf16x8*)(ar + 8 * half);
      af.h[1] = *(const bf16x8*)(ar + 16 + 8 * half);
    }
    // B fragments: lane column n = r16, K in table order (matches A)
    BF16x16 bf0, bf1;
    {
      const __bf16* br = &Bs[cur][0][32 * wn + r16];
      for (int s = 0; s < 16; ++s) {
        const int kk = (s < 8) ? (8 * half + s) : (16 + 8 * half + (s - 8));
        bf0.v[s] = br[kk * 72];
        bf1.v[s] = br[kk * 72 + 16];
      }
    }
    acc0 = wmma_bf16(af.v, bf0.v, acc0);
    acc1 = wmma_bf16(af.v, bf1.v, acc1);

    // Next buffer must be fully landed (and current fully consumed) before
    // the following iteration overwrites / reads buffers.
    wait_async0();
    __builtin_amdgcn_s_wait_tensorcnt(0);
    __syncthreads();
  }

  // Epilogue: C lane mapping -> col = l%16, row = p + 8*(l/16)
  for (int t = 0; t < 2; ++t) {
    f32x8 acc = t ? acc1 : acc0;
    const int col = n0 + 32 * wn + 16 * t + r16;
    if (col >= Nc) continue;
    for (int p = 0; p < 8; ++p) {
      const int row = m0 + 16 * wm + 8 * half + p;
      if (row >= M) continue;
      const float a = acc[p];
      if (epi == EPI_F32_RESID) {
        outF[(size_t)row * ldc + col] = a + resF[(size_t)row * ldc + col];
      } else if (epi == EPI_GELU_BF16) {
        const float g = 0.5f * a * (1.0f + erff(a * 0.70710678118654752f));
        outH[(size_t)row * ldc + col] = (__bf16)g;
      } else if (epi == EPI_KV) {
        const int b = row / Sseq, s = row % Sseq;
        if (col < cI) {
          const int h = col >> 6, dh = col & 63;
          kOut[(((size_t)b * cH + h) * cDH + dh) * Sseq + s] = (__bf16)a;      // kT: (B,H,DH,S)
        } else {
          const int c2 = col - cI, h = c2 >> 6, dh = c2 & 63;
          vOut[(((size_t)b * cH + h) * Sseq + s) * cDH + dh] = (__bf16)a;      // v : (B,H,S,DH)
        }
      } else { // EPI_QH
        const int b = row / cL, l = row % cL;
        const int h = col >> 6, dh = col & 63;
        outH[(((size_t)b * cH + h) * cL + l) * cDH + dh] = (__bf16)a;          // q : (B,H,L,DH)
      }
    }
  }
}

// ---------------------------------------------------------------------------
// Flash attention: grid = B*H blocks, 128 threads = 4 waves; wave w owns
// query rows [16w, 16w+16). Online softmax over S in chunks of 32.
// ---------------------------------------------------------------------------
__global__ __launch_bounds__(128) void attn_wmma(
    const __bf16* __restrict__ q,    // (B,H,L,DH)
    const __bf16* __restrict__ kT,   // (B,H,DH,S)
    const __bf16* __restrict__ v,    // (B,H,S,DH)
    const int*    __restrict__ mask, // (B,N)
    __bf16*       __restrict__ out)  // (B,L,I)
{
  const int bh   = blockIdx.x;
  const int b    = bh >> 3;
  const int wid  = threadIdx.x >> 5;
  const int lane = threadIdx.x & 31;
  const int half = lane >> 4;
  const int nloc = lane & 15;
  const int h    = bh & 7;

  __shared__ __bf16 P[4][16][40];   // per-wave 16x32 probability tile (padded)

  // Preload q A-fragments for K = DH = 64 (2 chunks of 32)
  const __bf16* qrow = q + (((size_t)bh) * cL + 16 * wid + nloc) * cDH;
  BF16x16 qf0, qf1;
  qf0.h[0] = *(const bf16x8*)(qrow + 8 * half);
  qf0.h[1] = *(const bf16x8*)(qrow + 16 + 8 * half);
  qf1.h[0] = *(const bf16x8*)(qrow + 32 + 8 * half);
  qf1.h[1] = *(const bf16x8*)(qrow + 32 + 16 + 8 * half);

  f32x8 O[4];
  for (int t = 0; t < 4; ++t) O[t] = fzero8();
  float Mx[8], Ls[8];
  for (int p = 0; p < 8; ++p) { Mx[p] = -1e30f; Ls[p] = 0.f; }

  const __bf16* kbase = kT + (size_t)bh * cDH * cS;
  const __bf16* vbase = v  + (size_t)bh * (size_t)cS * cDH;

  for (int s0 = 0; s0 < cS; s0 += 32) {
    // sim tiles for the two 16-col halves of this 32-key chunk
    f32x8 sim[2];
    for (int sc = 0; sc < 2; ++sc) {
      f32x8 a = fzero8();
      for (int c = 0; c < 2; ++c) {
        const __bf16* kp = kbase + (size_t)(32 * c) * cS + s0 + 16 * sc + nloc;
        BF16x16 bfr;
        for (int s = 0; s < 16; ++s) {
          const int kk = (s < 8) ? (8 * half + s) : (16 + 8 * half + (s - 8));
          bfr.v[s] = kp[(size_t)kk * cS];
        }
        a = wmma_bf16(c ? qf1.v : qf0.v, bfr.v, a);
      }
      sim[sc] = a;
    }

    // Mask lookup per column (fmask: real tokens get mask[b][s]; latents get 0)
    const int sA = s0 + nloc, sB = s0 + 16 + nloc;
    const bool mA = (sA < cN) && (mask[(size_t)b * cN + sA] == 1);
    const bool mB = (sB < cN) && (mask[(size_t)b * cN + sB] == 1);

    // Online softmax per row (rows p + 8*half; reduce over the 16-lane group)
    for (int p = 0; p < 8; ++p) {
      float x0 = mA ? -1e30f : sim[0][p] * 0.125f;   // scale = DH^-0.5
      float x1 = mB ? -1e30f : sim[1][p] * 0.125f;
      float rm = fmaxf(x0, x1);
      for (int d = 1; d < 16; d <<= 1) rm = fmaxf(rm, __shfl_xor(rm, d, 32));
      const float nm  = fmaxf(Mx[p], rm);
      const float scf = __expf(Mx[p] - nm);
      Mx[p] = nm;
      const float p0 = __expf(x0 - nm);
      const float p1 = __expf(x1 - nm);
      float rs = p0 + p1;
      for (int d = 1; d < 16; d <<= 1) rs += __shfl_xor(rs, d, 32);
      Ls[p] = Ls[p] * scf + rs;
      for (int t = 0; t < 4; ++t) O[t][p] *= scf;
      // Stage probabilities: row = p + 8*half, cols nloc and 16+nloc
      P[wid][8 * half + p][nloc]      = (__bf16)p0;
      P[wid][8 * half + p][16 + nloc] = (__bf16)p1;
    }

    // P as A-fragment (wave-local LDS; DS ops are in-order within a wave)
    BF16x16 pa;
    const __bf16* prow = &P[wid][nloc][0];
    pa.h[0] = *(const bf16x8*)(prow + 8 * half);
    pa.h[1] = *(const bf16x8*)(prow + 16 + 8 * half);

    // O += P(16x32) * v_chunk(32x64): 4 output n-tiles
    for (int t = 0; t < 4; ++t) {
      const __bf16* vp = vbase + (size_t)s0 * cDH + 16 * t + nloc;
      BF16x16 bfr;
      for (int s = 0; s < 16; ++s) {
        const int kk = (s < 8) ? (8 * half + s) : (16 + 8 * half + (s - 8));
        bfr.v[s] = vp[(size_t)kk * cDH];
      }
      O[t] = wmma_bf16(pa.v, bfr.v, O[t]);
    }
  }

  // Normalize and store: out[b, l, h*64 + dh]
  for (int p = 0; p < 8; ++p) {
    const float inv = 1.0f / Ls[p];
    const int lrow = 16 * wid + 8 * half + p;
    for (int t = 0; t < 4; ++t) {
      const float y = O[t][p] * inv;
      out[((size_t)b * cL + lrow) * cI + h * cDH + 16 * t + nloc] = (__bf16)y;
    }
  }
}

// ---------------------------------------------------------------------------
// LayerNorm: one wave per row (D = 1024 -> 32 elems/lane). Optional outputs:
// out1: bf16, row-mapped ((row/rn)*gs + row%rn + base); out2: bf16 compact;
// outF: fp32 compact (final LN).
// ---------------------------------------------------------------------------
__global__ __launch_bounds__(256) void ln_rows(
    const float* __restrict__ in, int R, int Dd,
    const float* __restrict__ w, const float* __restrict__ bb,
    __bf16* __restrict__ out1, int rn, long long gs, long long base,
    __bf16* __restrict__ out2, float* __restrict__ outF)
{
  const int row = blockIdx.x * 8 + (threadIdx.x >> 5);
  if (row >= R) return;
  const int lane = threadIdx.x & 31;
  const float* x = in + (size_t)row * Dd;

  float s = 0.f;
  for (int j = lane; j < Dd; j += 32) s += x[j];
  for (int d = 16; d; d >>= 1) s += __shfl_xor(s, d, 32);
  const float mu = s / (float)Dd;

  float var = 0.f;
  for (int j = lane; j < Dd; j += 32) { const float t = x[j] - mu; var += t * t; }
  for (int d = 16; d; d >>= 1) var += __shfl_xor(var, d, 32);
  const float rstd = rsqrtf(var / (float)Dd + 1e-5f);

  const size_t o1 = out1 ? ((size_t)((long long)(row / rn) * gs + (row % rn) + base) * Dd) : 0;
  for (int j = lane; j < Dd; j += 32) {
    const float y = (x[j] - mu) * rstd * w[j] + bb[j];
    if (out1) out1[o1 + j] = (__bf16)y;
    if (out2) out2[(size_t)row * Dd + j] = (__bf16)y;
    if (outF) outF[(size_t)row * Dd + j] = y;
  }
}

__global__ __launch_bounds__(256) void cast_f32_bf16(
    const float* __restrict__ in, __bf16* __restrict__ out, long long n)
{
  const long long i = (long long)blockIdx.x * blockDim.x + threadIdx.x;
  if (i < n) out[i] = (__bf16)in[i];
}

__global__ __launch_bounds__(256) void bcast_latents(
    const float* __restrict__ latents, float* __restrict__ lat, int total, int ld)
{
  const int i = blockIdx.x * blockDim.x + threadIdx.x;
  if (i < total) lat[i] = latents[i % ld];
}

// ---------------------------------------------------------------------------
extern "C" void kernel_launch(void* const* d_in, const int* in_sizes, int n_in,
                              void* d_out, int out_size, void* d_ws, size_t ws_size,
                              hipStream_t stream) {
  (void)in_sizes; (void)n_in; (void)out_size; (void)ws_size;
  const float* x        = (const float*)d_in[0];
  const int*   mask     = (const int*)  d_in[1];
  const float* latents  = (const float*)d_in[2];
  const float* ln_m_w   = (const float*)d_in[3];
  const float* ln_m_b   = (const float*)d_in[4];
  const float* ln_l_w   = (const float*)d_in[5];
  const float* ln_l_b   = (const float*)d_in[6];
  const float* Wq       = (const float*)d_in[7];
  const float* Wkv      = (const float*)d_in[8];
  const float* Wo       = (const float*)d_in[9];
  const float* ff_ln_w  = (const float*)d_in[10];
  const float* ff_ln_b  = (const float*)d_in[11];
  const float* W1       = (const float*)d_in[12];
  const float* W2       = (const float*)d_in[13];
  const float* norm_w   = (const float*)d_in[14];
  const float* norm_b   = (const float*)d_in[15];

  char* ws = (char*)d_ws;
  size_t off = 0;
  auto alloc = [&](size_t bytes) -> void* {
    void* p = ws + off;
    off = (off + bytes + 255) & ~(size_t)255;
    return p;
  };

  float*  lat    = (float*) alloc((size_t)cB * cL * cD * 4);
  __bf16* lnl_c  = (__bf16*)alloc((size_t)cB * cL * cD * 2);
  __bf16* lnff   = (__bf16*)alloc((size_t)cB * cL * cD * 2);
  __bf16* xseq   = (__bf16*)alloc((size_t)cB * cS * cD * 2);
  __bf16* qb     = (__bf16*)alloc((size_t)cB * cH * cL * cDH * 2);
  __bf16* kTb    = (__bf16*)alloc((size_t)cB * cH * cDH * cS * 2);
  __bf16* vb     = (__bf16*)alloc((size_t)cB * cH * cS * cDH * 2);
  __bf16* ao     = (__bf16*)alloc((size_t)cB * cL * cI * 2);
  __bf16* ffh    = (__bf16*)alloc((size_t)cB * cL * cFF * 2);
  __bf16* Wq_b   = (__bf16*)alloc((size_t)cDEPTH * cD * cI * 2);
  __bf16* Wkv_b  = (__bf16*)alloc((size_t)cDEPTH * cD * 2 * cI * 2);
  __bf16* Wo_b   = (__bf16*)alloc((size_t)cDEPTH * cI * cD * 2);
  __bf16* W1_b   = (__bf16*)alloc((size_t)cDEPTH * cD * cFF * 2);
  __bf16* W2_b   = (__bf16*)alloc((size_t)cDEPTH * cFF * cD * 2);

  // Weight casts (once per launch; deterministic)
  {
    const long long n1 = (long long)cDEPTH * cD * cI;
    const long long n2 = (long long)cDEPTH * cD * 2 * cI;
    const long long n3 = (long long)cDEPTH * cI * cD;
    const long long n4 = (long long)cDEPTH * cD * cFF;
    cast_f32_bf16<<<dim3((unsigned)((n1 + 255) / 256)), 256, 0, stream>>>(Wq,  Wq_b,  n1);
    cast_f32_bf16<<<dim3((unsigned)((n2 + 255) / 256)), 256, 0, stream>>>(Wkv, Wkv_b, n2);
    cast_f32_bf16<<<dim3((unsigned)((n3 + 255) / 256)), 256, 0, stream>>>(Wo,  Wo_b,  n3);
    cast_f32_bf16<<<dim3((unsigned)((n4 + 255) / 256)), 256, 0, stream>>>(W1,  W1_b,  n4);
    cast_f32_bf16<<<dim3((unsigned)((n4 + 255) / 256)), 256, 0, stream>>>(W2,  W2_b,  n4);
  }
  // lat = broadcast(latents) to (B, L, D), fp32
  {
    const int total = cB * cL * cD;
    bcast_latents<<<dim3((total + 255) / 256), 256, 0, stream>>>(latents, lat, total, cL * cD);
  }

  for (int i = 0; i < cDEPTH; ++i) {
    // xn = LN(x) -> xseq rows [b*S, b*S+N)
    ln_rows<<<dim3((cB * cN) / 8), 256, 0, stream>>>(
        x, cB * cN, cD, ln_m_w + (size_t)i * cD, ln_m_b + (size_t)i * cD,
        xseq, cN, (long long)cS, 0LL, nullptr, nullptr);
    // lnl = LN(lat) -> xseq tail rows [b*S+N, b*S+S) and compact lnl_c
    ln_rows<<<dim3((cB * cL) / 8), 256, 0, stream>>>(
        lat, cB * cL, cD, ln_l_w + (size_t)i * cD, ln_l_b + (size_t)i * cD,
        xseq, cL, (long long)cS, (long long)cN, lnl_c, nullptr);
    // q = lnl @ Wq  -> (B,H,L,DH)
    gemm_bf16_wmma<<<dim3(cI / 64, (cB * cL) / 64), 256, 0, stream>>>(
        lnl_c, cD, Wq_b + (size_t)i * cD * cI, cI,
        cB * cL, cI, cD, EPI_QH, cI,
        nullptr, nullptr, qb, nullptr, nullptr, 0);
    // kv = xseq @ Wkv -> kT (B,H,DH,S) and v (B,H,S,DH)
    gemm_bf16_wmma<<<dim3((2 * cI) / 64, (cB * cS) / 64), 256, 0, stream>>>(
        xseq, cD, Wkv_b + (size_t)i * cD * 2 * cI, 2 * cI,
        cB * cS, 2 * cI, cD, EPI_KV, 2 * cI,
        nullptr, nullptr, nullptr, kTb, vb, cS);
    // attention -> ao (B,L,I)
    attn_wmma<<<dim3(cB * cH), 128, 0, stream>>>(qb, kTb, vb, mask, ao);
    // lat = ao @ Wo + lat
    gemm_bf16_wmma<<<dim3(cD / 64, (cB * cL) / 64), 256, 0, stream>>>(
        ao, cI, Wo_b + (size_t)i * cI * cD, cD,
        cB * cL, cD, cI, EPI_F32_RESID, cD,
        lat, lat, nullptr, nullptr, nullptr, 0);
    // lnff = LN(lat)
    ln_rows<<<dim3((cB * cL) / 8), 256, 0, stream>>>(
        lat, cB * cL, cD, ff_ln_w + (size_t)i * cD, ff_ln_b + (size_t)i * cD,
        nullptr, 1, 0LL, 0LL, lnff, nullptr);
    // ffh = gelu(lnff @ W1)
    gemm_bf16_wmma<<<dim3(cFF / 64, (cB * cL) / 64), 256, 0, stream>>>(
        lnff, cD, W1_b + (size_t)i * cD * cFF, cFF,
        cB * cL, cFF, cD, EPI_GELU_BF16, cFF,
        nullptr, nullptr, ffh, nullptr, nullptr, 0);
    // lat = ffh @ W2 + lat
    gemm_bf16_wmma<<<dim3(cD / 64, (cB * cL) / 64), 256, 0, stream>>>(
        ffh, cFF, W2_b + (size_t)i * cFF * cD, cD,
        cB * cL, cD, cFF, EPI_F32_RESID, cD,
        lat, lat, nullptr, nullptr, nullptr, 0);
  }

  // Final LN -> fp32 output
  ln_rows<<<dim3((cB * cL) / 8), 256, 0, stream>>>(
      lat, cB * cL, cD, norm_w, norm_b,
      nullptr, 1, 0LL, 0LL, nullptr, (float*)d_out);
}
